// TransformerCritic_88837103551431
// MI455X (gfx1250) — matrix operand order repaired and checked by reading, not compiled, not run on hardware
//
#include <hip/hip_runtime.h>
#include <hip/hip_bf16.h>

typedef __attribute__((ext_vector_type(16))) _Float16 v16h;
typedef __attribute__((ext_vector_type(8)))  float    v8f;

#define OBS 294          // 7*7*6
#define NROWS 65536      // B*N
#define KPAD 320
#define WSA_COLS 299     // OBS + A

union Frag16 { v16h h; unsigned u[8]; };

// A-matrix fragment (16x32 f16, MxK). p points at row 0 of the 16-row tile
// (already offset to the K-chunk base); stride in halves.
__device__ static inline v16h frag_a(const _Float16* p, int stride, int m, int g) {
  Frag16 f;
  const unsigned* q = (const unsigned*)(p + m * stride);
#pragma unroll
  for (int v = 0; v < 8; ++v) {
    int k = ((v >= 4) ? 16 : 0) + ((v & 3) << 1) + (g << 3);  // even
    f.u[v] = q[k >> 1];
  }
  return f.h;
}

// B-matrix fragment (32x16 f16, KxN) sourced from o-major weights W[o][k].
__device__ static inline v16h frag_b(const _Float16* p, int stride, int n, int g) {
  Frag16 f;
  const unsigned* q = (const unsigned*)(p + n * stride);
#pragma unroll
  for (int v = 0; v < 8; ++v) f.u[v] = q[((v >= 4) ? 8 : 0) + (v & 3) + (g << 2)];
  return f.h;
}

__device__ static inline v8f wmma16(v16h a, v16h b, v8f c) {
  return __builtin_amdgcn_wmma_f32_16x16x32_f16(false, a, false, b, (short)0, c, false, false);
}

// ---------------- K0: one-shot f16 weight conversion into padded layouts
// cw16[320][320], wcat16[192][320], wqkv16[192][64], wf116[64][128]
__global__ __launch_bounds__(256) void k0_prep(const float* __restrict__ conv_w,
                                               const float* __restrict__ W_s,
                                               const float* __restrict__ W_sa,
                                               const float* __restrict__ W_e,
                                               const float* __restrict__ W_q,
                                               const float* __restrict__ W_k,
                                               const float* __restrict__ W_v,
                                               const float* __restrict__ W_f1,
                                               _Float16* __restrict__ cw16,
                                               _Float16* __restrict__ wcat16,
                                               _Float16* __restrict__ wqkv16,
                                               _Float16* __restrict__ wf116) {
  int i = blockIdx.x * 256 + threadIdx.x;
  if (i < 320 * 320) {
    int o = i / 320, k = i % 320;
    cw16[i] = (_Float16)((o < OBS && k < OBS) ? conv_w[(size_t)o * OBS + k] : 0.f);
  } else if (i < 320 * 320 + 192 * 320) {
    int j = i - 320 * 320;
    int o = j / 320, k = j % 320;
    float v = 0.f;
    if (k < OBS) {
      if (o < 64)       v = W_s[(size_t)o * OBS + k];
      else if (o < 128) v = W_sa[(size_t)(o - 64) * WSA_COLS + k];
      else              v = W_e[(size_t)(o - 128) * OBS + k];
    }
    wcat16[j] = (_Float16)v;
  } else if (i < 320 * 320 + 192 * 320 + 192 * 64) {
    int j = i - (320 * 320 + 192 * 320);
    int o = j >> 6, k = j & 63;
    const float* W = (o < 64) ? W_q : ((o < 128) ? W_k : W_v);
    wqkv16[j] = (_Float16)W[(size_t)(o & 63) * 64 + k];
  } else if (i < 320 * 320 + 192 * 320 + 192 * 64 + 64 * 128) {
    int j = i - (320 * 320 + 192 * 320 + 192 * 64);
    wf116[j] = (_Float16)W_f1[j];
  }
}

// ---------------- K1: conv-as-GEMM: s = states @ conv_w^T + conv_b + positions -> f16 (padded 320)
__global__ __launch_bounds__(256) void k1_conv(const float* __restrict__ states,
                                               const _Float16* __restrict__ cw16,
                                               const float* __restrict__ conv_b,
                                               const float* __restrict__ positions,
                                               _Float16* __restrict__ s16) {
  __shared__ _Float16 lds_a[2][64 * 32];
  __shared__ _Float16 lds_b[2][KPAD * 32];
  const int tid = threadIdx.x, lane = tid & 31, w = tid >> 5;
  const int rowbase = blockIdx.x * 64;
  const int mtile = (w & 3) * 16;
  const int nhalf = w >> 2;  // tiles [0..9] or [10..19]
  const int g = lane >> 4, ml = lane & 15;

  auto stage = [&](int kc, int buf) {
    const int kb = kc * 32;
    // A: 64 rows x 16 float2 segments (8-byte aligned: row stride 294*4)
    for (int i = tid; i < 64 * 16; i += 256) {
      int r = i >> 4, seg = i & 15;
      int k = kb + seg * 2;
      _Float16 h0 = (_Float16)0.f, h1 = (_Float16)0.f;
      if (k + 1 < OBS) {  // k even, OBS even -> full pair or nothing
        float2 v = *(const float2*)(states + (size_t)(rowbase + r) * OBS + k);
        h0 = (_Float16)v.x; h1 = (_Float16)v.y;
      }
      lds_a[buf][r * 32 + seg * 2]     = h0;
      lds_a[buf][r * 32 + seg * 2 + 1] = h1;
    }
    // B: straight b128 copies from pre-converted f16 weights
    for (int i = tid; i < KPAD * 4; i += 256) {
      int o = i >> 2, seg = i & 3;
      ((uint4*)lds_b[buf])[i] = *(const uint4*)(cw16 + (size_t)o * KPAD + kb + seg * 8);
    }
    if (kc + 1 < 10)  // hint next chunk of states into cache (global_prefetch_b8)
      __builtin_prefetch(states + (size_t)(rowbase + (tid & 63)) * OBS + kb + 32, 0, 1);
  };

  v8f z = {};
  v8f acc[10];
#pragma unroll
  for (int i = 0; i < 10; ++i) acc[i] = z;

  stage(0, 0);
  __syncthreads();
  for (int kc = 0; kc < 10; ++kc) {
    const int cur = kc & 1;
    if (kc < 9) stage(kc + 1, cur ^ 1);
    v16h a = frag_a(&lds_a[cur][mtile * 32], 32, ml, g);
#pragma unroll
    for (int t = 0; t < 10; ++t) {
      v16h b = frag_b(&lds_b[cur][(nhalf * 10 + t) * 16 * 32], 32, ml, g);
      acc[t] = wmma16(a, b, acc[t]);
    }
    __syncthreads();
  }
#pragma unroll
  for (int t = 0; t < 10; ++t) {
    int col = (nhalf * 10 + t) * 16 + ml;
#pragma unroll
    for (int i = 0; i < 8; ++i) {
      int row = rowbase + mtile + i + g * 8;
      float v = 0.f;
      if (col < OBS) v = acc[t][i] + conv_b[col] + positions[(row & 7) * OBS + col];
      s16[(size_t)row * KPAD + col] = (_Float16)v;
    }
  }
}

// ---------------- K2: E = lrelu([W_s|W_sa|W_e] GEMM + action term + bias) -> e16[row][192]
__global__ __launch_bounds__(256) void k2_feat(const _Float16* __restrict__ s16,
                                               const _Float16* __restrict__ wcat16,
                                               const float* __restrict__ W_sa,
                                               const float* __restrict__ b_s,
                                               const float* __restrict__ b_sa,
                                               const float* __restrict__ b_e,
                                               const float* __restrict__ actions,
                                               _Float16* __restrict__ e16) {
  __shared__ _Float16 lds_a[2][64 * 32];
  __shared__ _Float16 lds_b[2][192 * 32];
  const int tid = threadIdx.x, lane = tid & 31, w = tid >> 5;
  const int rowbase = blockIdx.x * 64;
  const int mtile = (w & 3) * 16;
  const int nhalf = w >> 2;  // 6 tiles each
  const int g = lane >> 4, ml = lane & 15;

  auto stage = [&](int kc, int buf) {
    const int kb = kc * 32;
    for (int i = tid; i < 64 * 4; i += 256) {  // A: b128 copies from padded s16
      int r = i >> 2, seg = i & 3;
      ((uint4*)lds_a[buf])[i] = *(const uint4*)(s16 + (size_t)(rowbase + r) * KPAD + kb + seg * 8);
    }
    for (int i = tid; i < 192 * 4; i += 256) {  // B: b128 copies from wcat16
      int o = i >> 2, seg = i & 3;
      ((uint4*)lds_b[buf])[i] = *(const uint4*)(wcat16 + (size_t)o * KPAD + kb + seg * 8);
    }
  };

  v8f z = {};
  v8f acc[6];
#pragma unroll
  for (int i = 0; i < 6; ++i) acc[i] = z;

  stage(0, 0);
  __syncthreads();
  for (int kc = 0; kc < 10; ++kc) {
    const int cur = kc & 1;
    if (kc < 9) stage(kc + 1, cur ^ 1);
    v16h a = frag_a(&lds_a[cur][mtile * 32], 32, ml, g);
#pragma unroll
    for (int t = 0; t < 6; ++t) {
      v16h b = frag_b(&lds_b[cur][(nhalf * 6 + t) * 16 * 32], 32, ml, g);
      acc[t] = wmma16(a, b, acc[t]);
    }
    __syncthreads();
  }
#pragma unroll
  for (int t = 0; t < 6; ++t) {
    int col = (nhalf * 6 + t) * 16 + ml;
#pragma unroll
    for (int i = 0; i < 8; ++i) {
      int row = rowbase + mtile + i + g * 8;
      float v = acc[t][i];
      if (col < 64) {
        v += b_s[col];
      } else if (col < 128) {
        int d = col - 64;
        v += b_sa[d];
        const float* ap = actions + (size_t)row * 5;  // row == b*8+n
#pragma unroll
        for (int j = 0; j < 5; ++j) v += W_sa[(size_t)d * WSA_COLS + OBS + j] * ap[j];
      } else {
        v += b_e[col - 128];
      }
      v = v > 0.f ? v : 0.01f * v;
      e16[(size_t)row * 192 + col] = (_Float16)v;
    }
  }
}

// ---------------- K3: [Q|K|V] GEMM from e16 (block-selected 64-wide inputs)
__global__ __launch_bounds__(256) void k3_qkv(const _Float16* __restrict__ e16,
                                              const _Float16* __restrict__ wqkv16,
                                              const float* __restrict__ b_q,
                                              const float* __restrict__ b_k,
                                              const float* __restrict__ b_v,
                                              _Float16* __restrict__ qkv16) {
  __shared__ _Float16 lds_a[64 * 192];
  __shared__ _Float16 lds_b[192 * 64];
  const int tid = threadIdx.x, lane = tid & 31, w = tid >> 5;
  const int rowbase = blockIdx.x * 64;
  const int mtile = (w & 3) * 16;
  const int nhalf = w >> 2;
  const int g = lane >> 4, ml = lane & 15;
  for (int i = tid; i < 64 * 192 / 8; i += 256)
    ((uint4*)lds_a)[i] = ((const uint4*)(e16 + (size_t)rowbase * 192))[i];
  for (int i = tid; i < 192 * 64 / 8; i += 256)
    ((uint4*)lds_b)[i] = ((const uint4*)wqkv16)[i];
  __syncthreads();
  v8f z = {};
#pragma unroll
  for (int t = 0; t < 6; ++t) {
    int nt = nhalf * 6 + t;
    int blk = nt >> 2;              // 0=Q, 1=K, 2=V
    int aoff = (blk == 0) ? 0 : 64; // Q reads e_s, K/V read e_sa
    v8f c = z;
#pragma unroll
    for (int kc = 0; kc < 2; ++kc) {
      v16h a = frag_a(lds_a + mtile * 192 + aoff + kc * 32, 192, ml, g);
      v16h b = frag_b(lds_b + (nt * 16) * 64 + kc * 32, 64, ml, g);
      c = wmma16(a, b, c);
    }
    int col = nt * 16 + ml;
    float bias = (col < 64) ? b_q[col] : ((col < 128) ? b_k[col - 64] : b_v[col - 128]);
#pragma unroll
    for (int i = 0; i < 8; ++i) {
      int row = rowbase + mtile + i + g * 8;
      qkv16[(size_t)row * 192 + col] = (_Float16)(c[i] + bias);
    }
  }
}

// ---------------- K4: attention scores, off-diagonal softmax (weight out), x = sum w*V
__global__ __launch_bounds__(256) void k4_attn(const _Float16* __restrict__ qkv16,
                                               float* __restrict__ out_w,
                                               _Float16* __restrict__ x16) {
  __shared__ float sc[8][8][8];
  __shared__ float ww[8][8][7];
  const int tid = threadIdx.x, lane = tid & 31, w = tid >> 5;
  const int b = blockIdx.x * 8 + w;
  const size_t base = (size_t)b * 8 * 192;
#pragma unroll
  for (int i = 0; i < 2; ++i) {
    int p = lane + 32 * i, n = p >> 3, m = p & 7;
    const _Float16* q = qkv16 + base + (size_t)n * 192;       // Q block
    const _Float16* k = qkv16 + base + (size_t)m * 192 + 64;  // K block
    float acc = 0.f;
#pragma unroll 8
    for (int d = 0; d < 64; ++d) acc += (float)q[d] * (float)k[d];
    sc[w][n][m] = acc * 0.125f;  // / sqrt(64)
  }
  __syncthreads();
  if (lane < 8) {
    int n = lane;
    float v[7], mx = -1e30f;
#pragma unroll
    for (int kk = 0; kk < 7; ++kk) {
      int m = kk < n ? kk : kk + 1;
      v[kk] = sc[w][n][m];
      mx = fmaxf(mx, v[kk]);
    }
    float s = 0.f;
#pragma unroll
    for (int kk = 0; kk < 7; ++kk) { v[kk] = __expf(v[kk] - mx); s += v[kk]; }
    float inv = 1.f / s;
#pragma unroll
    for (int kk = 0; kk < 7; ++kk) {
      float wv = v[kk] * inv;
      ww[w][n][kk] = wv;
      out_w[((size_t)b * 8 + n) * 7 + kk] = wv;
    }
  }
  __syncthreads();
  for (int n = 0; n < 8; ++n) {
#pragma unroll
    for (int j = 0; j < 2; ++j) {
      int d = lane + 32 * j;
      float x = 0.f;
#pragma unroll
      for (int kk = 0; kk < 7; ++kk) {
        int m = kk < n ? kk : kk + 1;
        x += ww[w][n][kk] * (float)qkv16[base + (size_t)m * 192 + 128 + d];
      }
      x16[((size_t)b * 8 + n) * 64 + d] = (_Float16)x;
    }
  }
}

// ---------------- K5: h = lrelu([e_q|x] @ W_f1^T + b_f1) via WMMA; value = h @ W_f2^T + b_f2
__global__ __launch_bounds__(256) void k5_out(const _Float16* __restrict__ e16,
                                              const _Float16* __restrict__ x16,
                                              const _Float16* __restrict__ wf116,
                                              const float* __restrict__ b_f1,
                                              const float* __restrict__ W_f2,
                                              const float* __restrict__ b_f2,
                                              float* __restrict__ out_v) {
  __shared__ _Float16 lds_a[64 * 128];
  __shared__ _Float16 lds_b[64 * 128];
  __shared__ _Float16 lds_h[64 * 64];
  const int tid = threadIdx.x, lane = tid & 31, w = tid >> 5;
  const int rowbase = blockIdx.x * 64;
  for (int i = tid; i < 64 * 16; i += 256) {  // A: [e_q | x], 16 uint4 per row
    int r = i >> 4, seg = i & 15;
    const _Float16* src = (seg < 8) ? (e16 + (size_t)(rowbase + r) * 192 + 128 + seg * 8)
                                    : (x16 + (size_t)(rowbase + r) * 64 + (seg - 8) * 8);
    ((uint4*)lds_a)[i] = *(const uint4*)src;
  }
  for (int i = tid; i < 64 * 128 / 8; i += 256) ((uint4*)lds_b)[i] = ((const uint4*)wf116)[i];
  __syncthreads();
  const int mtile = (w & 3) * 16, nhalf = w >> 2, g = lane >> 4, ml = lane & 15;
#pragma unroll
  for (int t = 0; t < 2; ++t) {
    int nt = nhalf * 2 + t;
    v8f c = {};
#pragma unroll
    for (int kc = 0; kc < 4; ++kc) {
      v16h a = frag_a(lds_a + mtile * 128 + kc * 32, 128, ml, g);
      v16h b = frag_b(lds_b + (nt * 16) * 128 + kc * 32, 128, ml, g);
      c = wmma16(a, b, c);
    }
    int col = nt * 16 + ml;
    float bias = b_f1[col];
#pragma unroll
    for (int i = 0; i < 8; ++i) {
      float h = c[i] + bias;
      h = h > 0.f ? h : 0.01f * h;
      lds_h[(mtile + i + g * 8) * 64 + col] = (_Float16)h;
    }
  }
  __syncthreads();
  for (int t2 = tid; t2 < 64 * 5; t2 += 256) {
    int r = t2 / 5, a = t2 % 5;
    float acc = b_f2[a];
#pragma unroll 8
    for (int d = 0; d < 64; ++d) acc += (float)lds_h[r * 64 + d] * W_f2[a * 64 + d];
    out_v[(size_t)(rowbase + r) * 5 + a] = acc;
  }
}

extern "C" void kernel_launch(void* const* d_in, const int* in_sizes, int n_in,
                              void* d_out, int out_size, void* d_ws, size_t ws_size,
                              hipStream_t stream) {
  const float* states    = (const float*)d_in[0];
  const float* actions   = (const float*)d_in[1];
  const float* conv_w    = (const float*)d_in[2];
  const float* conv_b    = (const float*)d_in[3];
  const float* positions = (const float*)d_in[4];
  const float* W_sa = (const float*)d_in[5];
  const float* b_sa = (const float*)d_in[6];
  const float* W_s  = (const float*)d_in[7];
  const float* b_s  = (const float*)d_in[8];
  const float* W_k  = (const float*)d_in[9];
  const float* b_k  = (const float*)d_in[10];
  const float* W_q  = (const float*)d_in[11];
  const float* b_q  = (const float*)d_in[12];
  const float* W_v  = (const float*)d_in[13];
  const float* b_v  = (const float*)d_in[14];
  const float* W_e  = (const float*)d_in[15];
  const float* b_e  = (const float*)d_in[16];
  const float* W_f1 = (const float*)d_in[17];
  const float* b_f1 = (const float*)d_in[18];
  const float* W_f2 = (const float*)d_in[19];
  const float* b_f2 = (const float*)d_in[20];

  char* ws = (char*)d_ws;
  size_t off = 0;
  _Float16* s16    = (_Float16*)(ws + off); off += (size_t)NROWS * KPAD * 2;
  _Float16* e16    = (_Float16*)(ws + off); off += (size_t)NROWS * 192 * 2;
  _Float16* qkv16  = (_Float16*)(ws + off); off += (size_t)NROWS * 192 * 2;
  _Float16* x16    = (_Float16*)(ws + off); off += (size_t)NROWS * 64 * 2;
  _Float16* cw16   = (_Float16*)(ws + off); off += (size_t)320 * 320 * 2;
  _Float16* wcat16 = (_Float16*)(ws + off); off += (size_t)192 * 320 * 2;
  _Float16* wqkv16 = (_Float16*)(ws + off); off += (size_t)192 * 64 * 2;
  _Float16* wf116  = (_Float16*)(ws + off); off += (size_t)64 * 128 * 2;

  float* out_v = (float*)d_out;         // B*N*A = 327680 floats
  float* out_w = out_v + 8192 * 8 * 5;  // B*N*7 = 458752 floats

  k0_prep<<<720, 256, 0, stream>>>(conv_w, W_s, W_sa, W_e, W_q, W_k, W_v, W_f1,
                                   cw16, wcat16, wqkv16, wf116);
  k1_conv<<<1024, 256, 0, stream>>>(states, cw16, conv_b, positions, s16);
  k2_feat<<<1024, 256, 0, stream>>>(s16, wcat16, W_sa, b_s, b_sa, b_e, actions, e16);
  k3_qkv <<<1024, 256, 0, stream>>>(e16, wqkv16, b_q, b_k, b_v, qkv16);
  k4_attn<<<1024, 256, 0, stream>>>(qkv16, out_w, x16);
  k5_out <<<1024, 256, 0, stream>>>(e16, x16, wf116, b_f1, W_f2, b_f2, out_v);
}
